// SeqAttention_59579786330678
// MI455X (gfx1250) — matrix operand order repaired and checked by reading, not compiled, not run on hardware
//
#include <hip/hip_runtime.h>

typedef __attribute__((ext_vector_type(16))) _Float16 v16h;
typedef __attribute__((ext_vector_type(8)))  _Float16 v8h;
typedef __attribute__((ext_vector_type(2)))  _Float16 v2h;
typedef __attribute__((ext_vector_type(8)))  float    v8f;

#define HEAD_DIM  128
#define N_HEADS   16
#define M_LEN     1024
#define SPAN_LEN  1024
#define RAMP      32.0f
#define LTILE     32

#define KL_STRIDE 136   // f16 per key row: 128 dims + 8 pad  (272B, 16B aligned, conflict-free)
#define VT_STRIDE 40    // f16 per dim row: 32 keys + 8 pad   (80B,  16B aligned, conflict-free)
#define P_STRIDE  40    // f16 per P row:   32 cols + 8 pad
#define KL_BUF    (32 * KL_STRIDE)
#define VT_BUF    (HEAD_DIM * VT_STRIDE)

// log2(e) / sqrt(128), pre-folded into Q before f16 conversion
#define SCL 0.12753653563184f

__device__ inline v8h cvt8(const float4 a, const float4 b) {
    v8h h;
    h[0] = (_Float16)a.x; h[1] = (_Float16)a.y; h[2] = (_Float16)a.z; h[3] = (_Float16)a.w;
    h[4] = (_Float16)b.x; h[5] = (_Float16)b.y; h[6] = (_Float16)b.z; h[7] = (_Float16)b.w;
    return h;
}

__device__ inline v16h join16(const v8h lo, const v8h hi) {
    v16h r;
#pragma unroll
    for (int i = 0; i < 8; ++i) { r[i] = lo[i]; r[i + 8] = hi[i]; }
    return r;
}

// ---- 16-lane reductions in pure VALU via DPP16 (no LDS traffic):
// xor1 = quad_perm(1,0,3,2)=0xB1, xor2 = quad_perm(2,3,0,1)=0x4E,
// then ring-combine the 4 uniform quads with row_ror:4 (0x124) and row_ror:8 (0x128).
template <int CTRL>
__device__ inline float dpp_xchg(float v) {
    int t = __builtin_amdgcn_update_dpp(0, __float_as_int(v), CTRL, 0xf, 0xf, true);
    return __int_as_float(t);
}
__device__ inline float redmax16(float v) {
    v = fmaxf(v, dpp_xchg<0xB1>(v));
    v = fmaxf(v, dpp_xchg<0x4E>(v));
    v = fmaxf(v, dpp_xchg<0x124>(v));
    v = fmaxf(v, dpp_xchg<0x128>(v));
    return v;
}
__device__ inline float redsum16(float v) {
    v += dpp_xchg<0xB1>(v);
    v += dpp_xchg<0x4E>(v);
    v += dpp_xchg<0x124>(v);
    v += dpp_xchg<0x128>(v);
    return v;
}

__global__ void __launch_bounds__(256)
adaptive_span_attn(const float* __restrict__ Q, const float* __restrict__ K,
                   const float* __restrict__ V, const float* __restrict__ spanv,
                   float* __restrict__ O)
{
    __shared__ _Float16 klds[2 * KL_BUF];            // K tiles, f16 [key][dim], double-buffered
    __shared__ _Float16 vtlds[2 * VT_BUF];           // V tiles, f16 [dim][key], double-buffered
    __shared__ _Float16 plds[8 * 16 * P_STRIDE];     // per-wave P scratch (C->A relayout)

    const int tid  = threadIdx.x;
    const int wave = tid >> 5;
    const int lane = tid & 31;
    const int half = lane >> 4;
    const int l16  = lane & 15;

    const int mblocks = M_LEN / 128;
    const int bh   = blockIdx.x / mblocks;
    const int mblk = blockIdx.x % mblocks;
    const int head = bh % N_HEADS;

    const float span = spanv[head];
    // mask > 0 iff j > 991 - span*1024 ; align start tile down (extra cols get e=0)
    float cut = 992.0f - span * (float)SPAN_LEN;
    int j0 = 0;
    if (cut > 0.0f) j0 = ((int)cut) & ~(LTILE - 1);
    const float spanC = span * (float)SPAN_LEN - ((float)SPAN_LEN - 1.0f - RAMP);
    const float invR  = 1.0f / RAMP;

    // ---- this wave's Q tile (16 x 128) -> A fragments, pre-scaled by log2e/sqrt(d)
    const int mbase = mblk * 128 + wave * 16;
    const float* qrow = Q + ((size_t)bh * M_LEN + (size_t)(mbase + l16)) * HEAD_DIM;
    v16h qa[4];
#pragma unroll
    for (int kc = 0; kc < 4; ++kc) {
        const float* g0 = qrow + kc * 32 + half * 8;  // K = {0..7|8..15}
        const float* g1 = g0 + 16;                    // K = {16..23|24..31}
        float4 a0 = ((const float4*)g0)[0], a1 = ((const float4*)g0)[1];
        float4 b0 = ((const float4*)g1)[0], b1 = ((const float4*)g1)[1];
        a0.x*=SCL; a0.y*=SCL; a0.z*=SCL; a0.w*=SCL;  a1.x*=SCL; a1.y*=SCL; a1.z*=SCL; a1.w*=SCL;
        b0.x*=SCL; b0.y*=SCL; b0.z*=SCL; b0.w*=SCL;  b1.x*=SCL; b1.y*=SCL; b1.z*=SCL; b1.w*=SCL;
        qa[kc] = join16(cvt8(a0, a1), cvt8(b0, b1));
    }

    v8f acc[8];
#pragma unroll
    for (int f = 0; f < 8; ++f) acc[f] = (v8f){0.f,0.f,0.f,0.f,0.f,0.f,0.f,0.f};
    float mrun[8], lrun[8];
#pragma unroll
    for (int c = 0; c < 8; ++c) { mrun[c] = -__builtin_inff(); lrun[c] = 0.0f; }

    const float* kbh = K + (size_t)bh * SPAN_LEN * HEAD_DIM;
    const float* vbh = V + (size_t)bh * SPAN_LEN * HEAD_DIM;
    _Float16* pw = plds + wave * 16 * P_STRIDE;

    // ---- staging thread mappings
    const int kk  = tid >> 3, dq = tid & 7;        // K: key, dim-group
    const int kp  = (tid & 15) * 2;                // V: key pair
    const int d0v = (tid >> 4) * 8;                // V: dim group

    // ---- software pipeline: global loads for tile jb+32 issued during compute of jb
    float4 kf0, kf1, kf2, kf3, va0, va1, vb0, vb1;
    auto load_tile = [&](int jb) {
        const float* src = kbh + ((size_t)(jb + kk)) * HEAD_DIM + dq * 16;
        kf0 = ((const float4*)src)[0]; kf1 = ((const float4*)src)[1];
        kf2 = ((const float4*)src)[2]; kf3 = ((const float4*)src)[3];
        const float* s0 = vbh + ((size_t)(jb + kp)) * HEAD_DIM + d0v;
        const float* s1 = s0 + HEAD_DIM;
        va0 = ((const float4*)s0)[0]; va1 = ((const float4*)s0)[1];
        vb0 = ((const float4*)s1)[0]; vb1 = ((const float4*)s1)[1];
    };

    load_tile(j0);
    int nbuf = 0;

    // incremental mask argument bases (avoid per-iteration int->float converts)
    float mb0 = (float)(j0 + l16) + spanC;
    float mb1 = mb0 + 16.0f;

    for (int jb = j0; jb < SPAN_LEN; jb += LTILE) {
        // ---- convert + store staged registers into LDS buffer `nbuf`
        {
            _Float16* dst = klds + nbuf * KL_BUF + kk * KL_STRIDE + dq * 16;
            *(v8h*)(dst)     = cvt8(kf0, kf1);
            *(v8h*)(dst + 8) = cvt8(kf2, kf3);
            _Float16* vt = vtlds + nbuf * VT_BUF;
            float av[8] = {va0.x,va0.y,va0.z,va0.w,va1.x,va1.y,va1.z,va1.w};
            float bw[8] = {vb0.x,vb0.y,vb0.z,vb0.w,vb1.x,vb1.y,vb1.z,vb1.w};
#pragma unroll
            for (int i = 0; i < 8; ++i) {
                v2h p; p[0] = (_Float16)av[i]; p[1] = (_Float16)bw[i];
                *(v2h*)(vt + (size_t)(d0v + i) * VT_STRIDE + kp) = p;
            }
        }
        __syncthreads();

        // ---- issue next tile's global loads (consumed at next iteration's store)
        if (jb + LTILE < SPAN_LEN) load_tile(jb + LTILE);
        if (jb + 2 * LTILE < SPAN_LEN) {
            const char* pk = (const char*)(kbh + (size_t)(jb + 2 * LTILE) * HEAD_DIM);
            const char* pv = (const char*)(vbh + (size_t)(jb + 2 * LTILE) * HEAD_DIM);
            __builtin_prefetch(pk + tid * 64, 0, 3);
            __builtin_prefetch(pv + tid * 64, 0, 3);
        }

        const _Float16* kl = klds + nbuf * KL_BUF;
        const _Float16* vt = vtlds + nbuf * VT_BUF;

        // ---- S = Q K^T, 2 key sub-tiles; split each accumulate chain into two
        //      independent depth-2 WMMA chains for XDL latency hiding
        v8f s[2];
#pragma unroll
        for (int t = 0; t < 2; ++t) {
            const _Float16* kr = kl + (size_t)(t * 16 + l16) * KL_STRIDE + half * 8;
            v16h b0 = join16(*(const v8h*)(kr),       *(const v8h*)(kr + 16));
            v16h b1 = join16(*(const v8h*)(kr + 32),  *(const v8h*)(kr + 48));
            v16h b2 = join16(*(const v8h*)(kr + 64),  *(const v8h*)(kr + 80));
            v16h b3 = join16(*(const v8h*)(kr + 96),  *(const v8h*)(kr + 112));
            v8f p0 = __builtin_amdgcn_wmma_f32_16x16x32_f16(false, qa[0], false, b0,
                     (short)0, (v8f){0.f,0.f,0.f,0.f,0.f,0.f,0.f,0.f}, false, false);
            v8f p1 = __builtin_amdgcn_wmma_f32_16x16x32_f16(false, qa[2], false, b2,
                     (short)0, (v8f){0.f,0.f,0.f,0.f,0.f,0.f,0.f,0.f}, false, false);
            p0 = __builtin_amdgcn_wmma_f32_16x16x32_f16(false, qa[1], false, b1,
                     (short)0, p0, false, false);
            p1 = __builtin_amdgcn_wmma_f32_16x16x32_f16(false, qa[3], false, b3,
                     (short)0, p1, false, false);
            s[t] = p0 + p1;
        }

        // ---- masked flash softmax (exp2 domain, scores pre-scaled), P -> LDS (f16)
        const float mask0 = fminf(fmaxf(mb0 * invR, 0.0f), 1.0f);
        const float mask1 = fminf(fmaxf(mb1 * invR, 0.0f), 1.0f);
        mb0 += (float)LTILE;
        mb1 += (float)LTILE;
#pragma unroll
        for (int c = 0; c < 8; ++c) {
            float a0 = s[0][c], a1 = s[1][c];
            float mx = redmax16(fmaxf(a0, a1));
            float mnew = fmaxf(mrun[c], mx);
            float corr = exp2f(mrun[c] - mnew);
            mrun[c] = mnew;
            float e0 = exp2f(a0 - mnew) * mask0;
            float e1 = exp2f(a1 - mnew) * mask1;
            float rs = redsum16(e0 + e1);
            lrun[c] = lrun[c] * corr + rs;
#pragma unroll
            for (int f = 0; f < 8; ++f) acc[f][c] *= corr;
            int row = c + half * 8;
            pw[row * P_STRIDE + l16]      = (_Float16)e0;
            pw[row * P_STRIDE + 16 + l16] = (_Float16)e1;
        }

        // ---- reload P in A-fragment layout (same-wave DS ordering; no barrier)
        const _Float16* pr = pw + l16 * P_STRIDE + half * 8;
        v16h pa = join16(*(const v8h*)pr, *(const v8h*)(pr + 16));

        // ---- O += P V  (8 independent WMMAs into distinct accumulators)
#pragma unroll
        for (int f = 0; f < 8; ++f) {
            const _Float16* vr = vt + (size_t)(f * 16 + l16) * VT_STRIDE + half * 8;
            v16h bb = join16(*(const v8h*)vr, *(const v8h*)(vr + 16));
            acc[f] = __builtin_amdgcn_wmma_f32_16x16x32_f16(false, pa, false, bb,
                                                            (short)0, acc[f], false, false);
        }
        nbuf ^= 1;
    }

    // ---- epilogue: divide by running sum, store f32 output
    float* orow = O + ((size_t)bh * M_LEN + (size_t)mbase) * HEAD_DIM;
#pragma unroll
    for (int c = 0; c < 8; ++c) {
        float inv = 1.0f / (lrun[c] + 1e-8f);
        int row = c + half * 8;
        float* dst = orow + (size_t)row * HEAD_DIM + l16;
#pragma unroll
        for (int f = 0; f < 8; ++f) dst[f * 16] = acc[f][c] * inv;
    }
}

extern "C" void kernel_launch(void* const* d_in, const int* in_sizes, int n_in,
                              void* d_out, int out_size, void* d_ws, size_t ws_size,
                              hipStream_t stream) {
    const float* Q = (const float*)d_in[0];
    const float* K = (const float*)d_in[1];
    const float* V = (const float*)d_in[2];
    const float* S = (const float*)d_in[3];
    float* O = (float*)d_out;
    int BH = in_sizes[0] / (M_LEN * HEAD_DIM);
    dim3 grid(BH * (M_LEN / 128));
    adaptive_span_attn<<<grid, dim3(256), 0, stream>>>(Q, K, V, S, O);
}
